// Head_31980326486123
// MI455X (gfx1250) — matrix operand order repaired and checked by reading, not compiled, not run on hardware
//
#include <hip/hip_runtime.h>

// Single-head causal attention, B=512 T=256 C=128 H=128, fp32 in/out.
// MI455X strategy: ~30 GFLOP vs ~270MB HBM traffic (~12us floor at 23.3TB/s),
// so all five GEMMs run on the WMMA pipe with bf16 operands / f32 accum
// (v_wmma_f32_16x16x32_bf16). bf16 copies of q,k,v live in d_ws (~96MB).
//
// Kernel 1: fused q/k/v projection GEMM. W^T staged bf16 in LDS so
//           B-fragments are contiguous ds_load_b128 pairs.
// Kernel 2: flash-attention per (batch, 128-row q tile): S=q@k^T via WMMA,
//           online softmax with wave32 shfl row reductions, P re-laid out
//           D->A through per-wave LDS (s_wait_dscnt 0 for the cross-lane
//           DS RAW), P@V via WMMA with V transposed into LDS.

typedef __attribute__((ext_vector_type(16))) __bf16 bf16x16;
typedef __attribute__((ext_vector_type(8)))  float  v8f;

union Frag16 {
  bf16x16 v;
  unsigned short u[16];
  uint4 q[2];
};

constexpr int Bb = 512, Tt = 256, Cc = 128, Hh = 128;

__device__ inline unsigned short f2bf(float f) {
  unsigned u = __builtin_bit_cast(unsigned, f);
  unsigned r = u + 0x7fffu + ((u >> 16) & 1u);   // round-to-nearest-even
  return (unsigned short)(r >> 16);
}

__device__ inline v8f zero8() {
  v8f z = {0.f, 0.f, 0.f, 0.f, 0.f, 0.f, 0.f, 0.f};
  return z;
}

// ---------------- Kernel 1: q/k/v projections ----------------
__global__ __launch_bounds__(256) void proj_kernel(
    const float* __restrict__ x,
    const float* __restrict__ Wk, const float* __restrict__ Wq,
    const float* __restrict__ Wv,
    float* __restrict__ k_out, float* __restrict__ v_out,
    unsigned short* __restrict__ qb, unsigned short* __restrict__ kb,
    unsigned short* __restrict__ vb) {
  __shared__ __align__(16) unsigned short wt[128][136];  // W^T bf16, padded

  const int tid  = threadIdx.x;
  const int wave = tid >> 5, lane = tid & 31;
  const int half = lane >> 4, col = lane & 15;
  const long m0 = (long)blockIdx.x * 128;

  // A fragments: 16 rows of x in bf16, contiguous K order
  // (lanes 0-15: K = 0..15; lanes 16-31: K = 16..31).
  Frag16 afrag[4];
  {
    const float* xr = x + (m0 + wave * 16 + col) * Cc;
#pragma unroll
    for (int ks = 0; ks < 4; ++ks) {
      const float* p0 = xr + ks * 32 + half * 16;
      const float* p1 = p0 + 8;
#pragma unroll
      for (int i = 0; i < 8; ++i) {
        afrag[ks].u[i]     = f2bf(p0[i]);
        afrag[ks].u[8 + i] = f2bf(p1[i]);
      }
    }
  }

  const float* Ws[3] = {Wk, Wq, Wv};
#pragma unroll
  for (int widx = 0; widx < 3; ++widx) {
    __syncthreads();  // previous wt consumers done before overwrite
    {                 // cooperative W^T staging (bf16)
      int kk = tid >> 1;
      int nb = (tid & 1) * 64;
      const float* src = Ws[widx] + kk * Hh + nb;
#pragma unroll
      for (int n = 0; n < 64; n += 4) {
        float4 f = *(const float4*)(src + n);
        wt[nb + n + 0][kk] = f2bf(f.x);
        wt[nb + n + 1][kk] = f2bf(f.y);
        wt[nb + n + 2][kk] = f2bf(f.z);
        wt[nb + n + 3][kk] = f2bf(f.w);
      }
    }
    __syncthreads();

    v8f acc[8];
#pragma unroll
    for (int nb = 0; nb < 8; ++nb) acc[nb] = zero8();

#pragma unroll
    for (int ks = 0; ks < 4; ++ks) {
#pragma unroll
      for (int nb = 0; nb < 8; ++nb) {
        Frag16 bfrag;
        const unsigned short* wr = &wt[nb * 16 + col][ks * 32 + half * 16];
        bfrag.q[0] = *(const uint4*)(wr);
        bfrag.q[1] = *(const uint4*)(wr + 8);
        acc[nb] = __builtin_amdgcn_wmma_f32_16x16x32_bf16(
            false, afrag[ks].v, false, bfrag.v, (short)0, acc[nb], false, false);
      }
    }

    float* fdst = (widx == 0) ? k_out : ((widx == 2) ? v_out : nullptr);
    unsigned short* bdst = (widx == 0) ? kb : ((widx == 1) ? qb : vb);
    const long rbase = m0 + wave * 16 + half * 8;   // D layout: M = r + 8*half
#pragma unroll
    for (int r = 0; r < 8; ++r) {
      const long row = rbase + r;
#pragma unroll
      for (int nb = 0; nb < 8; ++nb) {
        float val = acc[nb][r];
        long idx = row * Hh + nb * 16 + col;
        if (fdst) fdst[idx] = val;
        bdst[idx] = f2bf(val);
      }
    }
  }
}

// ---------------- Kernel 2: causal flash attention ----------------
__global__ __launch_bounds__(256) void attn_kernel(
    const unsigned short* __restrict__ qb, const unsigned short* __restrict__ kb,
    const unsigned short* __restrict__ vb, float* __restrict__ out) {
  __shared__ __align__(16) unsigned short vt[128][40];      // V^T tile, padded
  __shared__ __align__(16) unsigned short pbuf[8][16][32];  // per-wave P tile

  const int tid  = threadIdx.x;
  const int wave = tid >> 5, lane = tid & 31;
  const int half = lane >> 4, col = lane & 15;
  const int b  = blockIdx.x >> 1;
  const int q0 = (blockIdx.x & 1) * 128;
  const long base = (long)b * Tt;
  const float scale = 0.088388347648318447f;  // C^-0.5 (ref scales by n_embd)

  // Resident q A-fragments for this wave's 16 rows.
  Frag16 qfrag[4];
  {
    const unsigned short* qr = qb + (base + q0 + wave * 16 + col) * Hh;
#pragma unroll
    for (int ks = 0; ks < 4; ++ks) {
      qfrag[ks].q[0] = *(const uint4*)(qr + ks * 32 + half * 16);
      qfrag[ks].q[1] = *(const uint4*)(qr + ks * 32 + half * 16 + 8);
    }
  }

  v8f acc[8];
#pragma unroll
  for (int nb = 0; nb < 8; ++nb) acc[nb] = zero8();
  float m_r[8], l_r[8];
#pragma unroll
  for (int r = 0; r < 8; ++r) { m_r[r] = -1e30f; l_r[r] = 0.f; }

  const int jmax = q0 + 128;  // uniform causal bound for the whole workgroup
  for (int j = 0; j < jmax; j += 32) {
    __syncthreads();  // vt consumers of previous tile done
    {                 // cooperative V^T staging: 32 keys x 128 channels bf16
      int kk = tid >> 3;
      int nb = (tid & 7) * 16;
      const unsigned short* vr = vb + (base + j + kk) * Hh + nb;
      uint4 a = *(const uint4*)(vr);
      uint4 c = *(const uint4*)(vr + 8);
      unsigned w[8] = {a.x, a.y, a.z, a.w, c.x, c.y, c.z, c.w};
#pragma unroll
      for (int i = 0; i < 8; ++i) {
        vt[nb + 2 * i + 0][kk] = (unsigned short)(w[i] & 0xffffu);
        vt[nb + 2 * i + 1][kk] = (unsigned short)(w[i] >> 16);
      }
    }
    __syncthreads();

    // S = q @ k^T for two 16-key tiles (k rows read contiguous from global)
    v8f s0 = zero8(), s1 = zero8();
    const unsigned short* kr0 = kb + (base + j + col) * Hh;
    const unsigned short* kr1 = kb + (base + j + 16 + col) * Hh;
#pragma unroll
    for (int ks = 0; ks < 4; ++ks) {
      Frag16 kf0, kf1;
      kf0.q[0] = *(const uint4*)(kr0 + ks * 32 + half * 16);
      kf0.q[1] = *(const uint4*)(kr0 + ks * 32 + half * 16 + 8);
      kf1.q[0] = *(const uint4*)(kr1 + ks * 32 + half * 16);
      kf1.q[1] = *(const uint4*)(kr1 + ks * 32 + half * 16 + 8);
      s0 = __builtin_amdgcn_wmma_f32_16x16x32_bf16(
          false, qfrag[ks].v, false, kf0.v, (short)0, s0, false, false);
      s1 = __builtin_amdgcn_wmma_f32_16x16x32_bf16(
          false, qfrag[ks].v, false, kf1.v, (short)0, s1, false, false);
    }

    // Online softmax. This lane owns rows M = r + 8*half, column N = col.
    const int grow = q0 + wave * 16 + half * 8;
#pragma unroll
    for (int r = 0; r < 8; ++r) {
      const int row = grow + r;
      float v0 = s0[r] * scale;
      float v1 = s1[r] * scale;
      if (j + col > row)      v0 = -1e30f;
      if (j + 16 + col > row) v1 = -1e30f;
      float t = fmaxf(v0, v1);
#pragma unroll
      for (int off = 1; off < 16; off <<= 1)
        t = fmaxf(t, __shfl_xor(t, off, 32));
      const float mnew = fmaxf(m_r[r], t);
      float p0 = (v0 <= -9e29f) ? 0.f : __expf(v0 - mnew);
      float p1 = (v1 <= -9e29f) ? 0.f : __expf(v1 - mnew);
      const float alpha = __expf(m_r[r] - mnew);
      float ps = p0 + p1;
#pragma unroll
      for (int off = 1; off < 16; off <<= 1)
        ps += __shfl_xor(ps, off, 32);
      l_r[r] = l_r[r] * alpha + ps;
      m_r[r] = mnew;
      pbuf[wave][r + 8 * half][col]      = f2bf(p0);
      pbuf[wave][r + 8 * half][col + 16] = f2bf(p1);
#pragma unroll
      for (int nb = 0; nb < 8; ++nb) acc[nb][r] *= alpha;
    }

    // Cross-lane DS RAW: P written above (D layout) is re-read as A layout.
    asm volatile("s_wait_dscnt 0" ::: "memory");

    Frag16 pf;
    {
      const unsigned short* pr = &pbuf[wave][0][0];
      pf.q[0] = *(const uint4*)(pr + col * 32 + half * 16);
      pf.q[1] = *(const uint4*)(pr + col * 32 + half * 16 + 8);
    }
#pragma unroll
    for (int nb = 0; nb < 8; ++nb) {
      Frag16 vf;
      const unsigned short* vr = &vt[nb * 16 + col][half * 16];
      vf.q[0] = *(const uint4*)(vr);
      vf.q[1] = *(const uint4*)(vr + 8);
      acc[nb] = __builtin_amdgcn_wmma_f32_16x16x32_bf16(
          false, pf.v, false, vf.v, (short)0, acc[nb], false, false);
    }
  }

  // Finalize: divide by softmax denominator and store fp32.
#pragma unroll
  for (int r = 0; r < 8; ++r) {
    const float inv = 1.0f / l_r[r];
    const long row = base + q0 + wave * 16 + half * 8 + r;
#pragma unroll
    for (int nb = 0; nb < 8; ++nb)
      out[row * Hh + nb * 16 + col] = acc[nb][r] * inv;
  }
}

extern "C" void kernel_launch(void* const* d_in, const int* in_sizes, int n_in,
                              void* d_out, int out_size, void* d_ws, size_t ws_size,
                              hipStream_t stream) {
  const float* x  = (const float*)d_in[0];
  const float* Wk = (const float*)d_in[1];
  const float* Wq = (const float*)d_in[2];
  const float* Wv = (const float*)d_in[3];

  const size_t BTH = (size_t)Bb * Tt * Hh;   // 16,777,216
  float* out  = (float*)d_out;               // [B,T,H] attention output
  float* kout = out + BTH;                   // [B,T,H] k
  float* vout = out + 2 * BTH;               // [B,T,H] v

  // Workspace: bf16 q,k,v = 3 * BTH * 2 bytes (~96 MB)
  unsigned short* qb = (unsigned short*)d_ws;
  unsigned short* kb = qb + BTH;
  unsigned short* vb = qb + 2 * BTH;

  proj_kernel<<<(Bb * Tt) / 128, 256, 0, stream>>>(x, Wk, Wq, Wv,
                                                   kout, vout, qb, kb, vb);
  attn_kernel<<<Bb * (Tt / 128), 256, 0, stream>>>(qb, kb, vb, out);
}
